// SmoothedViterbi_61624190763040
// MI455X (gfx1250) — compile-verified, matrix operand order checked by reading, not run on hardware
//
#include <hip/hip_runtime.h>
#include <stdint.h>

// Smoothed Viterbi DP on MI455X (gfx1250).
// 16 persistent WGs x 256 threads. WG g owns rows [16g,16g+16); wave w owns
// columns [32w,32w+32) -> one v_wmma_f32_16x16x32_f16 per wave per step with
// A = exp(score-b) (f16) and B = ones, giving f32 row sums for logsumexp.
// Potentials stream HBM->LDS via global_load_async_to_lds_b128, 2 steps ahead.

#define T_LEN 4096
#define S_DIM 256
#define NWG 16
#define NTHREADS 256
#define NSLOT 3
#define ROWF 260            // padded LDS row stride (floats) to spread banks
#define ROWB (ROWF * 4)     // 1040 bytes, 16B aligned

typedef __attribute__((ext_vector_type(16))) _Float16 v16h;
typedef __attribute__((ext_vector_type(8)))  float    v8f;

// Low 32 bits of a generic pointer to __shared__ = LDS byte offset
// (LDS aperture lives entirely in addr[63:32] on CDNA5).
__device__ __forceinline__ uint32_t lds_off(const void* p) {
  return (uint32_t)(uintptr_t)p;
}

// Wave `wave` prefetches its own 16 rows x 32 cols window of pot[t] for
// workgroup wg into an LDS slot. 4 async instructions per wave per slot
// (each moves 16 B/lane = 512 B). Self-consistent: a wave only ever reads
// the columns it prefetched itself, so its own ASYNCcnt wait suffices.
__device__ __forceinline__ void async_slot_prefetch(const float* pot, int t, int wg,
                                                    const float* slotBase,
                                                    int wave, int lane) {
  uint32_t lanepart = (uint32_t)((lane & 7) * 16);
  uint32_t lbase = lds_off(slotBase) + (uint32_t)(128 * wave) + lanepart;
  uint64_t gbase = (uint64_t)(uintptr_t)pot
                 + (((uint64_t)t * 256ull + (uint64_t)wg * 16ull) * 256ull) * 4ull
                 + (uint64_t)(128 * wave);
  int r0 = lane >> 3;  // 0..3
#pragma unroll
  for (int i = 0; i < 4; ++i) {
    int r = 4 * i + r0;                                   // row 0..15
    uint32_t l = lbase + (uint32_t)(r * ROWB);            // LDS dest (16B aligned)
    uint32_t voff = (uint32_t)(r * 1024) + lanepart;      // global byte offset
    asm volatile("global_load_async_to_lds_b128 %0, %1, %2"
                 :: "v"(l), "v"(voff), "s"(gbase) : "memory");
  }
}

__device__ __forceinline__ void grid_barrier(unsigned* cnt, unsigned* gen) {
  __threadfence();          // every thread publishes its own stores
  __syncthreads();
  if (threadIdx.x == 0) {
    unsigned g = __hip_atomic_load(gen, __ATOMIC_SEQ_CST, __HIP_MEMORY_SCOPE_AGENT);
    unsigned a = __hip_atomic_fetch_add(cnt, 1u, __ATOMIC_SEQ_CST, __HIP_MEMORY_SCOPE_AGENT);
    if (a == NWG - 1) {
      __hip_atomic_store(cnt, 0u, __ATOMIC_SEQ_CST, __HIP_MEMORY_SCOPE_AGENT);
      __hip_atomic_fetch_add(gen, 1u, __ATOMIC_SEQ_CST, __HIP_MEMORY_SCOPE_AGENT);
    } else {
      while (__hip_atomic_load(gen, __ATOMIC_SEQ_CST, __HIP_MEMORY_SCOPE_AGENT) == g) {
        __builtin_amdgcn_s_sleep(2);
      }
    }
  }
  __syncthreads();
}

__global__ void init_barrier_kernel(unsigned* bar) {
  bar[0] = 0u;  // arrival count
  bar[1] = 0u;  // generation
}

__global__ __launch_bounds__(NTHREADS, 1)
void viterbi_kernel(const float* __restrict__ pot, float* __restrict__ out,
                    unsigned char* __restrict__ bp, unsigned* bar) {
  __shared__ __align__(16) float spot[NSLOT][16 * ROWF];  // ~49.9 KB
  __shared__ __align__(16) float vsh[S_DIM];
  __shared__ float psum[8][16];
  __shared__ float pmaxv[8][16];
  __shared__ int   pmaxi[8][16];
  __shared__ float bsh;

  const int tid  = threadIdx.x;
  const int wg   = blockIdx.x;
  const int wave = tid >> 5;
  const int lane = tid & 31;
  unsigned* cnt = bar;
  unsigned* gen = bar + 1;

  // v[0] = potentials[0, 0, :]
  if (wg == 0) out[T_LEN + tid] = pot[tid];

  // Prime the async pipeline: slots for t=1 and t=2.
  async_slot_prefetch(pot, 1, wg, &spot[1 % NSLOT][0], wave, lane);
  async_slot_prefetch(pot, 2, wg, &spot[2 % NSLOT][0], wave, lane);

  grid_barrier(cnt, gen);

  v16h bones;
#pragma unroll
  for (int i = 0; i < 16; ++i) bones[i] = (_Float16)1.0f;

  // CDNA5 f16 A-matrix 16x32 layout: lane m (hi=0) holds row m, K={0..7,16..23};
  // lane 16+m (hi=1) holds row m, K={8..15,24..31}. Wave's K-window = cols 32w..32w+31.
  const int m  = lane & 15;
  const int hi = lane >> 4;
  const int c0 = 32 * wave + 8 * hi;
  const int c1 = c0 + 16;

  for (int t = 1; t < T_LEN; ++t) {
    // Keep pipeline 2 steps deep (reuses the slot consumed at step t-1).
    if (t + 2 < T_LEN)
      async_slot_prefetch(pot, t + 2, wg, &spot[(t + 2) % NSLOT][0], wave, lane);

    // Stage v_prev into LDS (fresh global addresses each step).
    vsh[tid] = out[T_LEN + (t - 1) * S_DIM + tid];

    // Wait until slot t (this wave's own async loads) has landed in LDS.
    if (t + 2 < T_LEN)      asm volatile("s_wait_asynccnt 0x8" ::: "memory");
    else if (t + 1 < T_LEN) asm volatile("s_wait_asynccnt 0x4" ::: "memory");
    else                    asm volatile("s_wait_asynccnt 0x0" ::: "memory");
    __syncthreads();

    // b = max(v_prev) for stable logsumexp (wave 0 reduces).
    if (tid < 32) {
      float x = vsh[tid];
#pragma unroll
      for (int k = 1; k < 8; ++k) x = fmaxf(x, vsh[tid + 32 * k]);
#pragma unroll
      for (int o = 16; o >= 1; o >>= 1) x = fmaxf(x, __shfl_xor(x, o, 32));
      if (tid == 0) bsh = x;
    }
    __syncthreads();
    const float b = bsh;

    const float* rowp = &spot[t % NSLOT][m * ROWF];
    float4 pa0 = *(const float4*)(rowp + c0);
    float4 pa1 = *(const float4*)(rowp + c0 + 4);
    float4 pb0 = *(const float4*)(rowp + c1);
    float4 pb1 = *(const float4*)(rowp + c1 + 4);
    float4 va0 = *(const float4*)(&vsh[c0]);
    float4 va1 = *(const float4*)(&vsh[c0 + 4]);
    float4 vb0 = *(const float4*)(&vsh[c1]);
    float4 vb1 = *(const float4*)(&vsh[c1 + 4]);

    float mv = -3.0e38f;
    int   mi = 0;
    v16h a;
    // Ascending column order per lane => strict '>' keeps first argmax index.
    auto elem = [&](float pv, float vpv, int col, int ai) {
      float s = pv + vpv;
      if (s > mv) { mv = s; mi = col; }
      a[ai] = (_Float16)__expf(s - b);
    };
    elem(pa0.x, va0.x, c0 + 0, 0);  elem(pa0.y, va0.y, c0 + 1, 1);
    elem(pa0.z, va0.z, c0 + 2, 2);  elem(pa0.w, va0.w, c0 + 3, 3);
    elem(pa1.x, va1.x, c0 + 4, 4);  elem(pa1.y, va1.y, c0 + 5, 5);
    elem(pa1.z, va1.z, c0 + 6, 6);  elem(pa1.w, va1.w, c0 + 7, 7);
    elem(pb0.x, vb0.x, c1 + 0, 8);  elem(pb0.y, vb0.y, c1 + 1, 9);
    elem(pb0.z, vb0.z, c1 + 2, 10); elem(pb0.w, vb0.w, c1 + 3, 11);
    elem(pb1.x, vb1.x, c1 + 4, 12); elem(pb1.y, vb1.y, c1 + 5, 13);
    elem(pb1.z, vb1.z, c1 + 6, 14); elem(pb1.w, vb1.w, c1 + 7, 15);

    // Row sums over this wave's 32 columns: D = A x ones.
    v8f acc = {0.f, 0.f, 0.f, 0.f, 0.f, 0.f, 0.f, 0.f};
    v8f d = __builtin_amdgcn_wmma_f32_16x16x32_f16(
        false, a, false, bones, (short)0, acc, false, false);

    // Merge argmax across the two half-wave lanes covering each row.
    float ov = __shfl_xor(mv, 16, 32);
    int   oi = __shfl_xor(mi, 16, 32);
    if (ov > mv || (ov == mv && oi < mi)) { mv = ov; mi = oi; }
    if (hi == 0) { pmaxv[wave][m] = mv; pmaxi[wave][m] = mi; }

    // D layout: lanes 0-15 VGPR i = row i; lanes 16-31 VGPR i = row 8+i.
    if (lane == 0) {
#pragma unroll
      for (int i = 0; i < 8; ++i) psum[wave][i] = d[i];
    }
    if (lane == 16) {
#pragma unroll
      for (int i = 0; i < 8; ++i) psum[wave][8 + i] = d[i];
    }
    __syncthreads();

    if (tid < 16) {
      float ssum = 0.f;
#pragma unroll
      for (int w = 0; w < 8; ++w) ssum += psum[w][tid];
      float bv = pmaxv[0][tid];
      int   bi = pmaxi[0][tid];
#pragma unroll
      for (int w = 1; w < 8; ++w) {
        float cv = pmaxv[w][tid];
        int   ci = pmaxi[w][tid];
        if (cv > bv || (cv == bv && ci < bi)) { bv = cv; bi = ci; }
      }
      int grow = 16 * wg + tid;
      out[T_LEN + t * S_DIM + grow] = b + __logf(ssum);   // v_t
      bp[t * S_DIM + grow] = (unsigned char)bi;           // backpointer
    }
    grid_barrier(cnt, gen);
  }

  // Backtrace: argmax of v[T-1], then follow 1 MB (L2-resident) bp table.
  if (wg == 0 && tid == 0) {
    const float* vlast = &out[T_LEN + (T_LEN - 1) * S_DIM];
    float bv = vlast[0];
    int last = 0;
    for (int s = 1; s < S_DIM; ++s) {
      float v = vlast[s];
      if (v > bv) { bv = v; last = s; }
    }
    int s = last;
    out[T_LEN - 1] = (float)s;
    for (int t = T_LEN - 2; t >= 0; --t) {
      s = bp[(t + 1) * S_DIM + s];
      out[t] = (float)s;
    }
  }
}

extern "C" void kernel_launch(void* const* d_in, const int* in_sizes, int n_in,
                              void* d_out, int out_size, void* d_ws, size_t ws_size,
                              hipStream_t stream) {
  (void)in_sizes; (void)n_in; (void)out_size; (void)ws_size;
  const float* pot = (const float*)d_in[0];
  float* out = (float*)d_out;
  unsigned char* bp = (unsigned char*)d_ws;                       // 1 MB u8 backpointers
  unsigned* bar = (unsigned*)((char*)d_ws + (size_t)T_LEN * S_DIM);  // barrier state

  init_barrier_kernel<<<1, 1, 0, stream>>>(bar);
  viterbi_kernel<<<NWG, NTHREADS, 0, stream>>>(pot, out, bp, bar);
}
